// SNN_65927747994127
// MI455X (gfx1250) — matrix operand order repaired and checked by reading, not compiled, not run on hardware
//
#include <hip/hip_runtime.h>

// ---------------------------------------------------------------------------
// SNN forward, decoupled formulation:
//   cur1_all = X@W1^T + b1 + bias1   (one big GEMM, T*B=32768 x 2048, K=1024)
//   LIF1 recurrence per (b,h), parallel over 512K elements -> spk1 (d_out)
//   cur2_all = spk1@W2^T + b2        (GEMM 32768 x 10, K=2048, N padded to 16)
//   LIF2 recurrence per (b,n)        -> spk2 (d_out)
// Workspace: cur1 (268 MB f32) + cur2 (1.3 MB f32)  => ws >= ~270 MB.
// WMMA: v_wmma_f32_16x16x32_bf16 (spikes exact in bf16; f32 accumulate).
// gemm1 is a double-buffered pipeline: global loads of tile k+1 overlap the
// wmmas of tile k; one barrier per K-step.
// ---------------------------------------------------------------------------

typedef __bf16 v16bf __attribute__((ext_vector_type(16)));
typedef __bf16 v8bf  __attribute__((ext_vector_type(8)));
typedef float  v8f   __attribute__((ext_vector_type(8)));

#define SHUF16(lo, hi) __builtin_shufflevector(lo, hi, 0,1,2,3,4,5,6,7,8,9,10,11,12,13,14,15)

// LDS row padding: 40 bf16 = 80 B -> 16B-aligned b128 reads, bank-spread rows.
#define LDA 40

// ---------------- Kernel 1: cur1 = X @ W1^T + b1 + bias1 -------------------
// M=32768, N=2048, K=1024. Block tile 128x128x32, 256 threads = 8 waves (4x2).
__global__ __launch_bounds__(256) void snn_gemm1(
    const float* __restrict__ X,     // [32768, 1024]
    const float* __restrict__ W1,    // [2048, 1024]
    const float* __restrict__ b1,    // [2048]
    const float* __restrict__ bias1, // scalar
    float* __restrict__ cur1)        // [32768, 2048]
{
    const int K = 1024, N = 2048;
    __shared__ __bf16 sA[2][128 * LDA];   // 10 KB each
    __shared__ __bf16 sB[2][128 * LDA];

    const int tid   = threadIdx.x;
    const int lane  = tid & 31;
    const int wave  = tid >> 5;     // 0..7
    const int waveM = wave >> 1;    // 0..3  -> 32-row strip
    const int waveN = wave & 1;     // 0..1  -> 64-col strip
    const int r     = lane & 15;
    const int g     = lane >> 4;
    const int mBlock = blockIdx.y * 128;
    const int nBlock = blockIdx.x * 128;

    // cooperative loader: each thread owns 16 consecutive k of one row
    const int ldRow  = tid >> 1;          // 0..127
    const int ldHalf = (tid & 1) << 4;    // 0 or 16

    const float* gA = X  + (size_t)(mBlock + ldRow) * K + ldHalf;
    const float* gB = W1 + (size_t)(nBlock + ldRow) * K + ldHalf;

    v8f acc[2][4] = {};
    float4 ra[4], rb[4];                  // global->reg staging (next K tile)

    // prologue: fetch tile k0=0 and park it in LDS buffer 0
    #pragma unroll
    for (int i = 0; i < 4; i++) {
        ra[i] = *(const float4*)(gA + 4 * i);
        rb[i] = *(const float4*)(gB + 4 * i);
    }
    {
        __bf16* dA = &sA[0][ldRow * LDA + ldHalf];
        __bf16* dB = &sB[0][ldRow * LDA + ldHalf];
        #pragma unroll
        for (int i = 0; i < 4; i++) {
            dA[4*i+0] = (__bf16)ra[i].x; dA[4*i+1] = (__bf16)ra[i].y;
            dA[4*i+2] = (__bf16)ra[i].z; dA[4*i+3] = (__bf16)ra[i].w;
            dB[4*i+0] = (__bf16)rb[i].x; dB[4*i+1] = (__bf16)rb[i].y;
            dB[4*i+2] = (__bf16)rb[i].z; dB[4*i+3] = (__bf16)rb[i].w;
        }
    }
    __syncthreads();

    int buf = 0;
    for (int k0 = 0; k0 < K; k0 += 32) {
        const bool hasNext = (k0 + 32) < K;
        // issue next tile's global loads early (overlaps with wmma below)
        if (hasNext) {
            #pragma unroll
            for (int i = 0; i < 4; i++) {
                ra[i] = *(const float4*)(gA + (k0 + 32) + 4 * i);
                rb[i] = *(const float4*)(gB + (k0 + 32) + 4 * i);
            }
        }

        // ---- compute from LDS buffer `buf` ----
        // A fragments (16-bit A layout: g=0 -> K{0..7,16..23}; g=1 -> K{8..15,24..31})
        v16bf afrag[2];
        #pragma unroll
        for (int i = 0; i < 2; i++) {
            const int row = 32 * waveM + 16 * i + r;
            v8bf lo = *(const v8bf*)&sA[buf][row * LDA + 8 * g];
            v8bf hi = *(const v8bf*)&sA[buf][row * LDA + 16 + 8 * g];
            afrag[i] = SHUF16(lo, hi);
        }
        #pragma unroll
        for (int j = 0; j < 4; j++) {
            const int col = 64 * waveN + 16 * j + r;
            // B layout: lane (c, g) holds column c, K{8g..8g+7, 16+8g..16+8g+7}
            v8bf blo = *(const v8bf*)&sB[buf][col * LDA + 8 * g];
            v8bf bhi = *(const v8bf*)&sB[buf][col * LDA + 16 + 8 * g];
            v16bf bfrag = SHUF16(blo, bhi);
            #pragma unroll
            for (int i = 0; i < 2; i++)
                acc[i][j] = __builtin_amdgcn_wmma_f32_16x16x32_bf16(
                    false, afrag[i], false, bfrag, (short)0, acc[i][j], false, false);
        }

        // ---- park the prefetched tile in the other buffer ----
        if (hasNext) {
            __bf16* dA = &sA[buf ^ 1][ldRow * LDA + ldHalf];
            __bf16* dB = &sB[buf ^ 1][ldRow * LDA + ldHalf];
            #pragma unroll
            for (int i = 0; i < 4; i++) {
                dA[4*i+0] = (__bf16)ra[i].x; dA[4*i+1] = (__bf16)ra[i].y;
                dA[4*i+2] = (__bf16)ra[i].z; dA[4*i+3] = (__bf16)ra[i].w;
                dB[4*i+0] = (__bf16)rb[i].x; dB[4*i+1] = (__bf16)rb[i].y;
                dB[4*i+2] = (__bf16)rb[i].z; dB[4*i+3] = (__bf16)rb[i].w;
            }
        }
        __syncthreads();
        buf ^= 1;
    }

    // store (+ bias): C/D layout: lane l -> col = l&15, row = v + 8*(l>>4)
    const float bias = bias1[0];
    #pragma unroll
    for (int j = 0; j < 4; j++) {
        const int n  = nBlock + 64 * waveN + 16 * j + r;
        const float bb = b1[n] + bias;
        #pragma unroll
        for (int i = 0; i < 2; i++) {
            const int mBase = mBlock + 32 * waveM + 16 * i + 8 * g;
            #pragma unroll
            for (int v = 0; v < 8; v++)
                cur1[(size_t)(mBase + v) * N + n] = acc[i][j][v] + bb;
        }
    }
}

// ---------------- Kernel 2: LIF layer-1 recurrence -------------------------
// One thread per (b,h); sequential over T; writes spikes straight to d_out.
__global__ __launch_bounds__(256) void snn_lif1(
    const float* __restrict__ cur1,  // [T, B*nh]
    float* __restrict__ spk1,        // [T, B*nh] (d_out)
    int T)
{
    const size_t BN = (size_t)256 * 2048;
    const size_t idx = (size_t)blockIdx.x * 256 + threadIdx.x;
    float mem = 0.f;
    for (int t = 0; t < T; t++) {
        const float cur = cur1[(size_t)t * BN + idx];
        const float rst = (mem > 1.f) ? 1.f : 0.f;   // reset from previous mem
        mem = 0.9f * mem + cur - rst;                // BETA1=0.9, THR=1
        spk1[(size_t)t * BN + idx] = (mem > 1.f) ? 1.f : 0.f;
    }
}

// ---------------- Kernel 3: cur2 = spk1 @ W2^T + b2 ------------------------
// M=32768, K=2048, N=10 (padded to 16). 8 waves/block, one 16-row tile each.
#define LDW 1032   // padded bf16 row stride for half-K staging of W2
__global__ __launch_bounds__(256) void snn_gemm2(
    const float* __restrict__ spk1,  // [32768, 2048] (d_out)
    const float* __restrict__ W2,    // [10, 2048]
    const float* __restrict__ b2,    // [10]
    float* __restrict__ cur2)        // [32768, 10]
{
    const int K = 2048, NO = 10;
    __shared__ __bf16 sW[16 * LDW];  // 33 KB: 16 cols x half-K

    const int tid  = threadIdx.x;
    const int lane = tid & 31;
    const int wave = tid >> 5;
    const int r    = lane & 15;
    const int g    = lane >> 4;
    const int mBase = blockIdx.x * 128 + wave * 16;
    const int m     = mBase + r;     // A-matrix row for this lane

    v8f acc = {};
    for (int kh = 0; kh < 2; kh++) {
        // stage half of W2 (rows 0..9 real, 10..15 zero) as bf16
        for (int idx = tid; idx < 16 * 1024; idx += 256) {
            const int row = idx >> 10;
            const int kk  = idx & 1023;
            const float v = (row < NO) ? W2[row * K + kh * 1024 + kk] : 0.f;
            sW[row * LDW + kk] = (__bf16)v;
        }
        __syncthreads();

        for (int kk = 0; kk < 1024; kk += 32) {
            const float* pa = spk1 + (size_t)m * K + kh * 1024 + kk;
            // A frag: lane (r,g) -> K{8g..8g+7, 16+8g..16+8g+7}
            float4 f0 = *(const float4*)(pa + 8 * g);
            float4 f1 = *(const float4*)(pa + 8 * g + 4);
            float4 f2 = *(const float4*)(pa + 16 + 8 * g);
            float4 f3 = *(const float4*)(pa + 16 + 8 * g + 4);
            v16bf a;
            a[0]=(__bf16)f0.x; a[1]=(__bf16)f0.y; a[2]=(__bf16)f0.z; a[3]=(__bf16)f0.w;
            a[4]=(__bf16)f1.x; a[5]=(__bf16)f1.y; a[6]=(__bf16)f1.z; a[7]=(__bf16)f1.w;
            a[8]=(__bf16)f2.x; a[9]=(__bf16)f2.y; a[10]=(__bf16)f2.z; a[11]=(__bf16)f2.w;
            a[12]=(__bf16)f3.x; a[13]=(__bf16)f3.y; a[14]=(__bf16)f3.z; a[15]=(__bf16)f3.w;
            // B frag: lane (c=r, g) -> column c (= W2 row c), K{kk+8g..+7, kk+16+8g..+7}
            v8bf blo = *(const v8bf*)&sW[r * LDW + kk + 8 * g];
            v8bf bhi = *(const v8bf*)&sW[r * LDW + kk + 16 + 8 * g];
            v16bf b = SHUF16(blo, bhi);
            acc = __builtin_amdgcn_wmma_f32_16x16x32_bf16(
                false, a, false, b, (short)0, acc, false, false);
        }
        __syncthreads();
    }

    // store: lane col = r; only real outputs n < 10
    if (r < NO) {
        const float bb = b2[r];
        #pragma unroll
        for (int v = 0; v < 8; v++) {
            const int row = mBase + v + 8 * g;
            cur2[(size_t)row * NO + r] = acc[v] + bb;
        }
    }
}

// ---------------- Kernel 4: LIF layer-2 recurrence -------------------------
__global__ __launch_bounds__(256) void snn_lif2(
    const float* __restrict__ cur2,  // [T*256, 10]
    float* __restrict__ spk2,        // [T, 2560] (d_out tail)
    int T)
{
    const int idx = blockIdx.x * 256 + threadIdx.x;   // over B*no = 2560
    if (idx >= 2560) return;
    const int b = idx / 10, n = idx % 10;
    float mem = 0.f;
    for (int t = 0; t < T; t++) {
        const float cur = cur2[(size_t)(t * 256 + b) * 10 + n];
        const float rst = (mem > 1.f) ? 1.f : 0.f;
        mem = 0.8f * mem + cur - rst;                 // BETA2=0.8
        spk2[(size_t)t * 2560 + idx] = (mem > 1.f) ? 1.f : 0.f;
    }
}

// ---------------------------------------------------------------------------
extern "C" void kernel_launch(void* const* d_in, const int* in_sizes, int n_in,
                              void* d_out, int out_size, void* d_ws, size_t ws_size,
                              hipStream_t stream) {
    const float* x     = (const float*)d_in[0];  // [128, 256, 1024]
    const float* W1    = (const float*)d_in[1];  // [2048, 1024]
    const float* b1    = (const float*)d_in[2];  // [2048]
    const float* W2    = (const float*)d_in[3];  // [10, 2048]
    const float* b2    = (const float*)d_in[4];  // [10]
    const float* bias1 = (const float*)d_in[5];  // scalar
    const int T = 128;                            // d_in[6] holds T=128

    float* spk1 = (float*)d_out;                             // [128*256, 2048]
    float* spk2 = (float*)d_out + (size_t)128 * 256 * 2048;  // [128*256, 10]

    float* cur1 = (float*)d_ws;                              // 268 MB
    float* cur2 = (float*)((char*)d_ws + (size_t)32768 * 2048 * sizeof(float));

    dim3 g1(2048 / 128, 32768 / 128);   // (16, 256)  x-fastest -> L2 reuse of X
    snn_gemm1<<<g1, 256, 0, stream>>>(x, W1, b1, bias1, cur1);
    snn_lif1<<<(256 * 2048) / 256, 256, 0, stream>>>(cur1, spk1, T);
    snn_gemm2<<<32768 / 128, 256, 0, stream>>>(spk1, W2, b2, cur2);
    snn_lif2<<<(2560 + 255) / 256, 256, 0, stream>>>(cur2, spk2, T);
}
